// TFGNN_37099927503188
// MI455X (gfx1250) — matrix operand order repaired and checked by reading, not compiled, not run on hardware
//
#include <hip/hip_runtime.h>
#include <hip/hip_bf16.h>

typedef __attribute__((ext_vector_type(16))) __bf16 v16bf;
typedef __attribute__((ext_vector_type(8)))  float  v8f;

// ---------------- order-preserving float<->uint for atomic max ----------------
__device__ __forceinline__ unsigned fenc(float x) {
    unsigned u = __float_as_uint(x);
    return (u & 0x80000000u) ? ~u : (u | 0x80000000u);
}
__device__ __forceinline__ float fdec(unsigned u) {
    return (u & 0x80000000u) ? __uint_as_float(u & 0x7fffffffu) : __uint_as_float(~u);
}

// =============================================================================
// Fused multi-weight GEMM: Y_i[M x 128] = X[M x Kdim] @ W_i[Kdim x 128] + b_i,
// i = 0..nmat-1, bf16 WMMA with f32 accumulate.
// Block = 256 threads (8 waves), owns 128 rows. X tile staged to LDS ONCE,
// then reused for all nmat weight matrices (Q/K/V/root share the same X).
// =============================================================================
__global__ void __launch_bounds__(256)
tfgnn_gemm_wmma_multi(const float* __restrict__ X,
                      const float* __restrict__ W0, const float* __restrict__ W1,
                      const float* __restrict__ W2, const float* __restrict__ W3,
                      const float* __restrict__ b0, const float* __restrict__ b1,
                      const float* __restrict__ b2, const float* __restrict__ b3,
                      float* __restrict__ Y0, float* __restrict__ Y1,
                      float* __restrict__ Y2, float* __restrict__ Y3,
                      int M, int Kdim, int nmat) {
    __shared__ __bf16 sX [128][128];   // X tile, rows x K (zero padded)
    __shared__ __bf16 sWt[128][128];   // W transposed: sWt[n][k] = W[k][n]
    const float* Ws[4] = {W0, W1, W2, W3};
    const float* Bs[4] = {b0, b1, b2, b3};
    float*       Ys[4] = {Y0, Y1, Y2, Y3};

    const int tid = threadIdx.x;
    const int m0  = blockIdx.x * 128;

    // ---- stage X tile once (float4 global loads, convert to bf16) ----
    for (int idx = tid; idx < 128 * 32; idx += 256) {
        int row = idx >> 5;
        int k   = (idx & 31) * 4;
        int gm  = m0 + row;
        float4 v = make_float4(0.f, 0.f, 0.f, 0.f);
        if (gm < M && k < Kdim) v = *(const float4*)(X + (size_t)gm * Kdim + k);
        sX[row][k + 0] = (__bf16)v.x;
        sX[row][k + 1] = (__bf16)v.y;
        sX[row][k + 2] = (__bf16)v.z;
        sX[row][k + 3] = (__bf16)v.w;
    }

    const int wave    = tid >> 5;
    const int lane    = tid & 31;
    const int laneN   = lane & 15;
    const int rowA    = wave * 16 + laneN;      // A: M = lane&15
    const int khalfA  = (lane >> 4) * 8;        // A: K {0..7,16..23} / {8..15,24..31}
    const int khalfB  = (lane >> 4) * 8;        // B: K {0..7,16..23} / {8..15,24..31}
    const int nkt     = (Kdim + 31) >> 5;
    const int mo      = wave * 16 + (lane >> 4) * 8;  // D: lanes>=16 -> rows M=8..15

    for (int mat = 0; mat < nmat; ++mat) {
        __syncthreads();   // previous compute done before overwriting sWt
        const float* Wm = Ws[mat];
        for (int idx = tid; idx < 128 * 32; idx += 256) {
            int k = idx >> 5;
            int n = (idx & 31) * 4;
            float4 v = make_float4(0.f, 0.f, 0.f, 0.f);
            if (k < Kdim) v = *(const float4*)(Wm + (size_t)k * 128 + n);
            sWt[n + 0][k] = (__bf16)v.x;
            sWt[n + 1][k] = (__bf16)v.y;
            sWt[n + 2][k] = (__bf16)v.z;
            sWt[n + 3][k] = (__bf16)v.w;
        }
        __syncthreads();   // sWt (and, for mat==0, sX) visible

        v8f acc[8] = {};
        for (int kt = 0; kt < nkt; ++kt) {
            const int ka = kt * 32 + khalfA;
            v16bf a;
#pragma unroll
            for (int i = 0; i < 8; ++i) { a[i] = sX[rowA][ka + i]; a[8 + i] = sX[rowA][ka + 16 + i]; }
            const int kb = kt * 32 + khalfB;
#pragma unroll
            for (int nt = 0; nt < 8; ++nt) {
                const int col = nt * 16 + laneN;
                v16bf b;
#pragma unroll
                for (int i = 0; i < 8; ++i) { b[i] = sWt[col][kb + i]; b[8 + i] = sWt[col][kb + 16 + i]; }
                acc[nt] = __builtin_amdgcn_wmma_f32_16x16x32_bf16(
                    false, a, false, b, (short)0, acc[nt], false, false);
            }
        }

        const float* bias = Bs[mat];
        float*       Y    = Ys[mat];
#pragma unroll
        for (int nt = 0; nt < 8; ++nt) {
            const int col = nt * 16 + laneN;
            const float bv = bias[col];
#pragma unroll
            for (int j = 0; j < 8; ++j) {
                const int gm = m0 + mo + j;
                if (gm < M) Y[(size_t)gm * 128 + col] = acc[nt][j] + bv;
            }
        }
    }
}

// r[n][f] = sum_d We[f][d] * q[n][d]   (factor q.e = ea.(We q))
__global__ void tfgnn_qr_kernel(const float* __restrict__ q, const float* __restrict__ We,
                                float* __restrict__ r, int N) {
    int idx = blockIdx.x * blockDim.x + threadIdx.x;
    if (idx >= N * 16) return;
    int n = idx >> 4, f = idx & 15;
    const float* qn = q + (size_t)n * 128;
    const float* wf = We + f * 128;
    float acc = 0.0f;
#pragma unroll 8
    for (int d = 0; d < 128; ++d) acc += qn[d] * wf[d];
    r[idx] = acc;
}

// Pass 1: score per edge + segment max via atomic uint-max. One wave per edge.
__global__ void tfgnn_edge_score(const float* __restrict__ q, const float* __restrict__ k,
                                 const float* __restrict__ r, const float* __restrict__ ea,
                                 const int* __restrict__ src, const int* __restrict__ dst,
                                 float* __restrict__ s, unsigned* __restrict__ m, int E) {
    int e = (blockIdx.x * blockDim.x + threadIdx.x) >> 5;
    int lane = threadIdx.x & 31;
    if (e >= E) return;
    int sn = src[e], dn = dst[e];
    const float* qd = q + (size_t)dn * 128;
    const float* ks = k + (size_t)sn * 128;
    float acc = 0.0f;
#pragma unroll
    for (int i = 0; i < 4; ++i) { int d = lane + 32 * i; acc += qd[d] * ks[d]; }
    if (lane < 16) acc += ea[(size_t)e * 16 + lane] * r[(size_t)dn * 16 + lane];
    for (int off = 16; off > 0; off >>= 1) acc += __shfl_xor(acc, off, 32);
    float score = acc * 0.08838834764831845f;   // 1/sqrt(128)
    if (lane == 0) {
        s[e] = score;
        atomicMax(m + dn, fenc(score));
    }
}

// Pass 2: w=exp(s-m); recompute e = ea@We (We in LDS); atomic-add sum and w*(v[src]+e).
__global__ void __launch_bounds__(256)
tfgnn_edge_agg(const float* __restrict__ v, const float* __restrict__ ea,
               const float* __restrict__ We, const int* __restrict__ src,
               const int* __restrict__ dst, const float* __restrict__ s,
               const unsigned* __restrict__ m, float* __restrict__ ssum,
               float* __restrict__ agg, int E) {
    __shared__ float sWe[16 * 128];
    for (int i = threadIdx.x; i < 16 * 128; i += blockDim.x) sWe[i] = We[i];
    __syncthreads();
    int e = (blockIdx.x * blockDim.x + threadIdx.x) >> 5;
    int lane = threadIdx.x & 31;
    if (e >= E) return;
    int sn = src[e], dn = dst[e];
    float w = __expf(s[e] - fdec(m[dn]));
    if (lane == 0) atomicAdd(ssum + dn, w);
    float eav[16];
#pragma unroll
    for (int f = 0; f < 16; ++f) eav[f] = ea[(size_t)e * 16 + f];
#pragma unroll
    for (int i = 0; i < 4; ++i) {
        int d = lane + 32 * i;
        float ev = 0.0f;
#pragma unroll
        for (int f = 0; f < 16; ++f) ev += eav[f] * sWe[f * 128 + d];
        atomicAdd(agg + (size_t)dn * 128 + d, w * (v[(size_t)sn * 128 + d] + ev));
    }
}

// Pass 3: root += agg / (sum + 1e-16)
__global__ void tfgnn_combine(float* __restrict__ root, const float* __restrict__ agg,
                              const float* __restrict__ ssum, long total) {
    long i = (long)blockIdx.x * blockDim.x + threadIdx.x;
    if (i >= total) return;
    long n = i >> 7;
    root[i] += agg[i] / (ssum[n] + 1e-16f);
}

// Graph LayerNorm: two-level deterministic reduction, then affine + ReLU.
__global__ void tfgnn_reduce1(const float* __restrict__ x, float* __restrict__ partials,
                              long total) {
    __shared__ float ss[256], sq[256];
    float a = 0.0f, b = 0.0f;
    for (long i = (long)blockIdx.x * blockDim.x + threadIdx.x; i < total;
         i += (long)gridDim.x * blockDim.x) {
        float v = x[i]; a += v; b += v * v;
    }
    ss[threadIdx.x] = a; sq[threadIdx.x] = b;
    __syncthreads();
    for (int o = 128; o > 0; o >>= 1) {
        if ((int)threadIdx.x < o) { ss[threadIdx.x] += ss[threadIdx.x + o]; sq[threadIdx.x] += sq[threadIdx.x + o]; }
        __syncthreads();
    }
    if (threadIdx.x == 0) { partials[2 * blockIdx.x] = ss[0]; partials[2 * blockIdx.x + 1] = sq[0]; }
}

__global__ void tfgnn_reduce2(const float* __restrict__ partials, int nparts,
                              float* __restrict__ stats, float inv_total) {
    __shared__ float ss[256], sq[256];
    float a = 0.0f, b = 0.0f;
    for (int i = threadIdx.x; i < nparts; i += 256) { a += partials[2 * i]; b += partials[2 * i + 1]; }
    ss[threadIdx.x] = a; sq[threadIdx.x] = b;
    __syncthreads();
    for (int o = 128; o > 0; o >>= 1) {
        if ((int)threadIdx.x < o) { ss[threadIdx.x] += ss[threadIdx.x + o]; sq[threadIdx.x] += sq[threadIdx.x + o]; }
        __syncthreads();
    }
    if (threadIdx.x == 0) {
        float mean = ss[0] * inv_total;
        float var  = sq[0] * inv_total - mean * mean;
        stats[0] = mean;
        stats[1] = rsqrtf(var + 1e-5f);
    }
}

__global__ void tfgnn_ln_relu(const float* __restrict__ x, const float* __restrict__ stats,
                              const float* __restrict__ w, const float* __restrict__ b,
                              float* __restrict__ y, long total) {
    long i = (long)blockIdx.x * blockDim.x + threadIdx.x;
    if (i >= total) return;
    int d = (int)(i & 127);
    float v = (x[i] - stats[0]) * stats[1] * w[d] + b[d];
    y[i] = v > 0.0f ? v : 0.0f;
}

// out[n] = h[n] . fc2_w + fc2_b   (one wave per node)
__global__ void tfgnn_final_fc(const float* __restrict__ h, const float* __restrict__ w,
                               const float* __restrict__ b, float* __restrict__ out, int N) {
    int n = (blockIdx.x * blockDim.x + threadIdx.x) >> 5;
    int lane = threadIdx.x & 31;
    if (n >= N) return;
    float acc = 0.0f;
#pragma unroll
    for (int i = 0; i < 4; ++i) { int d = lane + 32 * i; acc += h[(size_t)n * 128 + d] * w[d]; }
    for (int off = 16; off > 0; off >>= 1) acc += __shfl_xor(acc, off, 32);
    if (lane == 0) out[n] = acc + b[0];
}

// =============================================================================
extern "C" void kernel_launch(void* const* d_in, const int* in_sizes, int n_in,
                              void* d_out, int out_size, void* d_ws, size_t ws_size,
                              hipStream_t stream) {
    const float* x     = (const float*)d_in[0];
    const int*   ei    = (const int*)  d_in[1];
    const float* ea    = (const float*)d_in[2];
    const float* fc1_w = (const float*)d_in[3];
    const float* fc1_b = (const float*)d_in[4];
    const float* W_[2][4] = {{(const float*)d_in[5],  (const float*)d_in[7],
                              (const float*)d_in[9],  (const float*)d_in[12]},
                             {(const float*)d_in[16], (const float*)d_in[18],
                              (const float*)d_in[20], (const float*)d_in[23]}};
    const float* B_[2][4] = {{(const float*)d_in[6],  (const float*)d_in[8],
                              (const float*)d_in[10], (const float*)d_in[13]},
                             {(const float*)d_in[17], (const float*)d_in[19],
                              (const float*)d_in[21], (const float*)d_in[24]}};
    const float* We_[2]  = {(const float*)d_in[11], (const float*)d_in[22]};
    const float* lnw_[2] = {(const float*)d_in[14], (const float*)d_in[25]};
    const float* lnb_[2] = {(const float*)d_in[15], (const float*)d_in[26]};
    const float* fc2_w   = (const float*)d_in[27];
    const float* fc2_b   = (const float*)d_in[28];

    const int N = in_sizes[0] / 64;
    const int E = in_sizes[1] / 2;
    const int* srcp = ei;
    const int* dstp = ei + E;

    // ---- workspace layout ----
    char* p = (char*)d_ws;
    auto wsalloc = [&](size_t bytes) -> void* {
        void* r = (void*)p;
        p += (bytes + 255) & ~(size_t)255;
        return r;
    };
    float*    hx    = (float*)wsalloc((size_t)N * 128 * 4);  // layer input / ln output
    float*    qb    = (float*)wsalloc((size_t)N * 128 * 4);
    float*    kb    = (float*)wsalloc((size_t)N * 128 * 4);
    float*    vb    = (float*)wsalloc((size_t)N * 128 * 4);
    float*    root  = (float*)wsalloc((size_t)N * 128 * 4);  // skip term, becomes pre-LN h
    float*    agg   = (float*)wsalloc((size_t)N * 128 * 4);
    float*    rb    = (float*)wsalloc((size_t)N * 16 * 4);
    float*    sedge = (float*)wsalloc((size_t)E * 4);
    float*    ssum  = (float*)wsalloc((size_t)N * 4);
    unsigned* mmax  = (unsigned*)wsalloc((size_t)N * 4);
    float*    parts = (float*)wsalloc(2 * 1024 * 4);
    float*    stats = (float*)wsalloc(2 * 4);
    (void)ws_size; (void)n_in; (void)out_size;

    const long totNH  = (long)N * 128;
    const int  nbRows = (N + 127) / 128;
    const int  nbNH   = (int)((totNH + 255) / 256);
    const int  nbEdge = (E + 7) / 8;              // 8 waves per 256-thread block
    const int  nbNode = (N + 7) / 8;

    // fc1: hx = x @ fc1_w + fc1_b   (K = 64, single matrix)
    tfgnn_gemm_wmma_multi<<<nbRows, 256, 0, stream>>>(
        x, fc1_w, fc1_w, fc1_w, fc1_w, fc1_b, fc1_b, fc1_b, fc1_b,
        hx, hx, hx, hx, N, 64, 1);

    for (int L = 0; L < 2; ++L) {
        // fused Q/K/V/root GEMMs (K = 128): X staged once, 4 weight matrices
        tfgnn_gemm_wmma_multi<<<nbRows, 256, 0, stream>>>(
            hx, W_[L][0], W_[L][1], W_[L][2], W_[L][3],
            B_[L][0], B_[L][1], B_[L][2], B_[L][3],
            qb, kb, vb, root, N, 128, 4);
        // r = We @ q per node
        tfgnn_qr_kernel<<<(N * 16 + 255) / 256, 256, 0, stream>>>(qb, We_[L], rb, N);
        // reset per-call accumulators
        hipMemsetAsync(mmax, 0, (size_t)N * 4, stream);
        hipMemsetAsync(ssum, 0, (size_t)N * 4, stream);
        hipMemsetAsync(agg,  0, (size_t)N * 128 * 4, stream);
        // segment softmax (3 passes)
        tfgnn_edge_score<<<nbEdge, 256, 0, stream>>>(qb, kb, rb, ea, srcp, dstp, sedge, mmax, E);
        tfgnn_edge_agg  <<<nbEdge, 256, 0, stream>>>(vb, ea, We_[L], srcp, dstp, sedge, mmax,
                                                     ssum, agg, E);
        tfgnn_combine<<<nbNH, 256, 0, stream>>>(root, agg, ssum, totNH);
        // graph layernorm + relu -> hx (next layer input)
        tfgnn_reduce1<<<1024, 256, 0, stream>>>(root, parts, totNH);
        tfgnn_reduce2<<<1, 256, 0, stream>>>(parts, 1024, stats, 1.0f / (float)totNH);
        tfgnn_ln_relu<<<nbNH, 256, 0, stream>>>(root, stats, lnw_[L], lnb_[L], hx, totNH);
    }

    // fc2: out = h @ fc2_w + fc2_b
    tfgnn_final_fc<<<nbNode, 256, 0, stream>>>(hx, fc2_w, fc2_b, (float*)d_out, N);
}
